// AdaptiveSoftmaxProbe_13245679141252
// MI455X (gfx1250) — matrix operand order repaired and checked by reading, not compiled, not run on hardware
//
#include <hip/hip_runtime.h>
#include <hip/hip_bf16.h>
#include <stdint.h>

// ---------------------------------------------------------------------------
// AdaptiveLogSoftmaxWithLoss on gfx1250 (MI455X), bf16 WMMA path.
//   head  = x @ Wh^T          [4096 x 1002]   (materialized, padded to 1024)
//   h0    = x @ W1_0^T        [4096 x 512]    (bf16)
//   h1    = x @ W1_1^T        [4096 x 128]    (bf16)
//   tail logits are NEVER materialized: fused WMMA + online logsumexp.
// ---------------------------------------------------------------------------

typedef __attribute__((ext_vector_type(8)))  float   v8f;
typedef __attribute__((ext_vector_type(8)))  __bf16  v8bf;
typedef __attribute__((ext_vector_type(16))) __bf16  v16bf;

#define NEG_INF (-__builtin_inff())

static __device__ __forceinline__ v16bf make_a_frag(const __bf16* p) {
  // A 16x32 bf16 layout: lane m=L&15; lanes hold K in {kb..kb+15}
  // with kb = 16*(L>>4). Elements 0..7 contiguous at +0, 8..15 contiguous at +8.
  v8bf lo = *(const v8bf*)(p);
  v8bf hi = *(const v8bf*)(p + 8);
  return __builtin_shufflevector(lo, hi, 0,1,2,3,4,5,6,7,8,9,10,11,12,13,14,15);
}

static __device__ __forceinline__ v8f wmma_bf16(v16bf a, v16bf b, v8f c) {
  return __builtin_amdgcn_wmma_f32_16x16x32_bf16(false, a, false, b, (short)0, c,
                                                 false, false);
}

// ---------------------------------------------------------------------------
// fp32 -> bf16 conversion with zero row-padding (flat element space).
// ---------------------------------------------------------------------------
__global__ void cvt_bf16_pad(const float* __restrict__ src, __bf16* __restrict__ dst,
                             int srcElems, int dstElems) {
  int i = blockIdx.x * blockDim.x + threadIdx.x;
  if (i < dstElems) dst[i] = (i < srcElems) ? (__bf16)src[i] : (__bf16)0.0f;
}

// ---------------------------------------------------------------------------
// Generic bf16 GEMM: C[M x ldc] = A[M x K] * B[Npad x K]^T.
// Wave tile: 16 rows x 64 cols (4 x v8f accumulators). block = 128 thr = 4 waves.
// ---------------------------------------------------------------------------
__global__ void wmma_gemm_bf16(const __bf16* __restrict__ A, const __bf16* __restrict__ B,
                               float* __restrict__ Cf, __bf16* __restrict__ Cb,
                               int M, int Npad, int K, int ldc) {
  const int lane = threadIdx.x & 31;
  const int wave = threadIdx.x >> 5;
  const int gw   = blockIdx.x * (blockDim.x >> 5) + wave;
  const int ntiles = Npad >> 6;                 // 64-wide wave tiles in N
  const int mt = gw / ntiles;
  const int nt = gw % ntiles;
  if (mt * 16 >= M) return;

  const int m  = lane & 15;
  const int hs = lane >> 4;
  const int kb = hs * 16;                       // A K-group base
  const int ks = hs * 16;                       // B K-group base
  const int n0 = nt * 64 + m;                   // this lane's column in subtile 0

  const __bf16* Ap = A + (size_t)(mt * 16 + m) * K + kb;
  const __bf16* Bp = B + (size_t)n0 * K + ks;
  const size_t bstride = (size_t)16 * K;        // 16 columns forward

  v8f acc0 = {}, acc1 = {}, acc2 = {}, acc3 = {};
  for (int k0 = 0; k0 < K; k0 += 32) {
    v16bf a  = make_a_frag(Ap + k0);
    v16bf b0 = *(const v16bf*)(Bp + 0 * bstride + k0);
    v16bf b1 = *(const v16bf*)(Bp + 1 * bstride + k0);
    v16bf b2 = *(const v16bf*)(Bp + 2 * bstride + k0);
    v16bf b3 = *(const v16bf*)(Bp + 3 * bstride + k0);
    if (k0 + 256 < K) __builtin_prefetch(Bp + k0 + 256, 0, 1);
    acc0 = wmma_bf16(a, b0, acc0);
    acc1 = wmma_bf16(a, b1, acc1);
    acc2 = wmma_bf16(a, b2, acc2);
    acc3 = wmma_bf16(a, b3, acc3);
  }

  // C/D layout: lane col n = L&15, VGPR v -> row m = v + 8*(L>>4)
  const int rbase = mt * 16 + 8 * hs;
  const int cbase = nt * 64 + m;
#pragma unroll
  for (int v = 0; v < 8; ++v) {
    const size_t r = (size_t)(rbase + v) * ldc;
    if (Cf) {
      Cf[r + cbase +  0] = acc0[v];
      Cf[r + cbase + 16] = acc1[v];
      Cf[r + cbase + 32] = acc2[v];
      Cf[r + cbase + 48] = acc3[v];
    } else {
      Cb[r + cbase +  0] = (__bf16)acc0[v];
      Cb[r + cbase + 16] = (__bf16)acc1[v];
      Cb[r + cbase + 32] = (__bf16)acc2[v];
      Cb[r + cbase + 48] = (__bf16)acc3[v];
    }
  }
}

// ---------------------------------------------------------------------------
// Head stats: per row, logsumexp over 1002 real columns + target/cluster gathers.
// One wave per row; block = 256 (8 waves).
// ---------------------------------------------------------------------------
__global__ void head_stats(const float* __restrict__ head, const int* __restrict__ target,
                           float* __restrict__ lseH, float* __restrict__ headT,
                           float* __restrict__ c0o, float* __restrict__ c1o) {
  const int lane = threadIdx.x & 31;
  const int row  = blockIdx.x * (blockDim.x >> 5) + (threadIdx.x >> 5);
  if (row >= 4096) return;
  const float* hp = head + (size_t)row * 1024;

  float mx = NEG_INF;
  for (int c = lane; c < 1002; c += 32) mx = fmaxf(mx, hp[c]);
  for (int off = 1; off < 32; off <<= 1) mx = fmaxf(mx, __shfl_xor(mx, off, 32));
  float s = 0.0f;
  for (int c = lane; c < 1002; c += 32) s += __expf(hp[c] - mx);
  for (int off = 1; off < 32; off <<= 1) s += __shfl_xor(s, off, 32);

  if (lane == 0) {
    lseH[row] = mx + __logf(s);
    int t = target[row];
    t = t < 0 ? 0 : (t > 999 ? 999 : t);
    headT[row] = hp[t];
    c0o[row] = hp[1000];
    c1o[row] = hp[1001];
  }
}

// ---------------------------------------------------------------------------
// Fused tail: for a 16-row stripe, sweep all Npad classes in 128-col steps
// (8 waves x one 16x16 WMMA tile), keep per-lane online (max,sum) softmax
// state, grab the target logit in flight. Merge lanes/waves at stripe end.
// ---------------------------------------------------------------------------
template <int K>
__global__ void tail_fused(const __bf16* __restrict__ Hb, const __bf16* __restrict__ W2b,
                           const int* __restrict__ target,
                           float* __restrict__ lse_out, float* __restrict__ g_out,
                           int Npad, int osz, int low) {
  __shared__ __attribute__((aligned(16))) __bf16 sh[16 * K];
  __shared__ float smax[8][16];
  __shared__ float ssum[8][16];

  const int tid  = threadIdx.x;
  const int lane = tid & 31;
  const int wave = tid >> 5;
  const int stripe = blockIdx.x;
  const int m  = lane & 15;
  const int hs = lane >> 4;
  const int kb = hs * 16;
  const int ks = hs * 16;

  // Stage the h stripe (16 x K bf16) in LDS.
  for (int i = tid * 8; i < 16 * K; i += blockDim.x * 8)
    *(v8bf*)(&sh[i]) = *(const v8bf*)(Hb + (size_t)stripe * 16 * K + i);
  __syncthreads();

  float rm[8], rs[8];
  int   tt[8];
#pragma unroll
  for (int v = 0; v < 8; ++v) {
    rm[v] = NEG_INF; rs[v] = 0.0f;
    tt[v] = target[stripe * 16 + v + 8 * hs] - low;   // may be out of range: never matches
  }

  const __bf16* Apl = sh + m * K + kb;
  for (int cb = wave * 16; cb < Npad; cb += 128) {
    const int col = cb + m;
    const __bf16* Bp = W2b + (size_t)col * K + ks;
    v8f acc = {};
#pragma unroll
    for (int k0 = 0; k0 < K; k0 += 32) {
      v16bf a = make_a_frag(Apl + k0);
      v16bf b = *(const v16bf*)(Bp + k0);
      acc = wmma_bf16(a, b, acc);
    }
    const bool valid = (col < osz);
#pragma unroll
    for (int v = 0; v < 8; ++v) {
      const float val = acc[v];
      if (tt[v] == col) g_out[stripe * 16 + v + 8 * hs] = val;  // target logit
      if (valid) {                                              // online logsumexp
        const float nm = fmaxf(rm[v], val);
        rs[v] = rs[v] * __expf(rm[v] - nm) + __expf(val - nm);
        rm[v] = nm;
      }
    }
  }

  // Merge across the 16 lanes of each half-wave (rows v+8*hs).
#pragma unroll
  for (int v = 0; v < 8; ++v) {
    for (int off = 1; off < 16; off <<= 1) {
      const float om = __shfl_xor(rm[v], off, 32);
      const float os = __shfl_xor(rs[v], off, 32);
      const float nm = fmaxf(rm[v], om);
      float acc_s = (rm[v] > NEG_INF) ? rs[v] * __expf(rm[v] - nm) : 0.0f;
      if (om > NEG_INF) acc_s += os * __expf(om - nm);
      rs[v] = acc_s;
      rm[v] = nm;
    }
  }
  if (m == 0) {
#pragma unroll
    for (int v = 0; v < 8; ++v) { smax[wave][v + 8 * hs] = rm[v]; ssum[wave][v + 8 * hs] = rs[v]; }
  }
  __syncthreads();
  if (tid < 16) {
    float nm = NEG_INF;
    for (int w = 0; w < 8; ++w) nm = fmaxf(nm, smax[w][tid]);
    float s = 0.0f;
    for (int w = 0; w < 8; ++w)
      if (smax[w][tid] > NEG_INF) s += ssum[w][tid] * __expf(smax[w][tid] - nm);
    lse_out[stripe * 16 + tid] = nm + __logf(s);
  }
}

// ---------------------------------------------------------------------------
// Combine per-token log-prob + deterministic partial reduction for the loss.
// ---------------------------------------------------------------------------
__global__ void combine_out(const int* __restrict__ target,
                            const float* __restrict__ lseH, const float* __restrict__ headT,
                            const float* __restrict__ c0, const float* __restrict__ c1,
                            const float* __restrict__ lse0, const float* __restrict__ g0,
                            const float* __restrict__ lse1, const float* __restrict__ g1,
                            float* __restrict__ out, float* __restrict__ partial) {
  __shared__ float red[256];
  const int i = blockIdx.x * blockDim.x + threadIdx.x;
  const int t = target[i];
  const float lh = lseH[i];
  float o;
  if (t < 1000)        o = headT[i] - lh;
  else if (t < 10000)  o = (c0[i] - lh) + (g0[i] - lse0[i]);
  else                 o = (c1[i] - lh) + (g1[i] - lse1[i]);
  out[i] = o;
  red[threadIdx.x] = o;
  __syncthreads();
  for (int s = 128; s > 0; s >>= 1) {
    if (threadIdx.x < s) red[threadIdx.x] += red[threadIdx.x + s];
    __syncthreads();
  }
  if (threadIdx.x == 0) partial[blockIdx.x] = red[0];
}

__global__ void finalize_loss(const float* __restrict__ partial, float* __restrict__ loss) {
  if (threadIdx.x == 0) {
    float s = 0.0f;
    for (int b = 0; b < 16; ++b) s += partial[b];
    *loss = -s / 4096.0f;
  }
}

// ---------------------------------------------------------------------------
extern "C" void kernel_launch(void* const* d_in, const int* in_sizes, int n_in,
                              void* d_out, int out_size, void* d_ws, size_t ws_size,
                              hipStream_t stream) {
  const float* x      = (const float*)d_in[0];   // [4096, 2048]
  const int*   target = (const int*)  d_in[1];   // [4096]
  const float* W_head = (const float*)d_in[2];   // [1002, 2048]
  const float* W1_0   = (const float*)d_in[3];   // [512, 2048]
  const float* W2_0   = (const float*)d_in[4];   // [9000, 512]
  const float* W1_1   = (const float*)d_in[5];   // [128, 2048]
  const float* W2_1   = (const float*)d_in[6];   // [40257, 128]
  float* out = (float*)d_out;                    // [4096] + [1] loss

  const int M = 4096, D = 2048;
  const int HPAD = 1024;                 // 1002 -> 1024
  const int N0 = 9000,  N0P = 9216;      // tail0 classes, padded to 128-mult
  const int N1 = 40257, N1P = 40448;     // tail1 classes, padded to 128-mult

  uintptr_t p = (uintptr_t)d_ws;
  auto carve = [&](size_t bytes) { uintptr_t q = p; p = (p + bytes + 255) & ~(uintptr_t)255; return (void*)q; };

  __bf16* xb   = (__bf16*)carve((size_t)M * D * 2);
  __bf16* Whb  = (__bf16*)carve((size_t)HPAD * D * 2);
  __bf16* W10b = (__bf16*)carve((size_t)512 * D * 2);
  __bf16* W11b = (__bf16*)carve((size_t)128 * D * 2);
  __bf16* W20b = (__bf16*)carve((size_t)N0P * 512 * 2);
  __bf16* W21b = (__bf16*)carve((size_t)N1P * 128 * 2);
  float*  head = (float*) carve((size_t)M * HPAD * 4);
  __bf16* h0b  = (__bf16*)carve((size_t)M * 512 * 2);
  __bf16* h1b  = (__bf16*)carve((size_t)M * 128 * 2);
  float* lseH  = (float*)carve(M * 4);
  float* headT = (float*)carve(M * 4);
  float* c0    = (float*)carve(M * 4);
  float* c1    = (float*)carve(M * 4);
  float* lse0  = (float*)carve(M * 4);
  float* g0    = (float*)carve(M * 4);
  float* lse1  = (float*)carve(M * 4);
  float* g1    = (float*)carve(M * 4);
  float* part  = (float*)carve(64 * 4);

  auto cvt = [&](const float* s, __bf16* d, int se, int de) {
    cvt_bf16_pad<<<(de + 255) / 256, 256, 0, stream>>>(s, d, se, de);
  };
  cvt(x,      xb,   M * D,        M * D);
  cvt(W_head, Whb,  1002 * D,     HPAD * D);
  cvt(W1_0,   W10b, 512 * D,      512 * D);
  cvt(W1_1,   W11b, 128 * D,      128 * D);
  cvt(W2_0,   W20b, N0 * 512,     N0P * 512);
  cvt(W2_1,   W21b, N1 * 128,     N1P * 128);

  // GEMMs: waveTiles = (M/16) * (Npad/64); 4 waves per block.
  wmma_gemm_bf16<<<(M / 16) * (HPAD / 64) / 4, 128, 0, stream>>>(xb, Whb,  head, nullptr, M, HPAD, D, HPAD);
  wmma_gemm_bf16<<<(M / 16) * (512  / 64) / 4, 128, 0, stream>>>(xb, W10b, nullptr, h0b,  M, 512,  D, 512);
  wmma_gemm_bf16<<<(M / 16) * (128  / 64) / 4, 128, 0, stream>>>(xb, W11b, nullptr, h1b,  M, 128,  D, 128);

  head_stats<<<M / 8, 256, 0, stream>>>(head, target, lseH, headT, c0, c1);

  tail_fused<512><<<M / 16, 256, 0, stream>>>(h0b, W20b, target, lse0, g0, N0P, N0, 1000);
  tail_fused<128><<<M / 16, 256, 0, stream>>>(h1b, W21b, target, lse1, g1, N1P, N1, 10000);

  combine_out<<<16, 256, 0, stream>>>(target, lseH, headT, c0, c1, lse0, g0, lse1, g1, out, part);
  finalize_loss<<<1, 32, 0, stream>>>(part, out + 4096);
}
